// PointNetSetAbstractionPytorch3D_81733227643084
// MI455X (gfx1250) — matrix operand order repaired and checked by reading, not compiled, not run on hardware
//
#include <hip/hip_runtime.h>

// ---------------------------------------------------------------------------
// Problem constants (from reference)
// ---------------------------------------------------------------------------
#define BATCH   8
#define NPTS    8192
#define CIN     64
#define SCENT   2048      // NPOINT
#define KNB     32        // NSAMPLE
#define C1      64
#define C2      128
#define C3      256

// LDS strides (in halves). Row stride in dwords mod 64 has gcd 4 => 16
// consecutive rows land in 16 distinct banks; row bytes % 16 == 0.
#define ST_X   104        // gathered features, 96 cols used (67 real + pad)
#define ST_Y   72         // layer1 out (64 cols)
#define ST_Z   136        // layer2 out (128 cols)
#define ST_W1  72         // W1^T: 96 rows x 64 cols
#define ST_W2  136        // W2^T: 64 rows x 128 cols
#define ST_W3  264        // W3^T: 128 rows x 256 cols

// Dynamic LDS layout (bytes, all 16B aligned)
#define OFF_BUFX   0
#define OFF_BUFY   26624                    // 128*ST_X*2
#define OFF_BUFZ   (26624+18432)            // +128*ST_Y*2
#define OFF_WBUF   (45056+34816)            // +128*ST_Z*2      = 79872
#define OFF_SB     (79872+67584)            // +128*ST_W3*2     = 147456
#define OFF_POOL   (147456+3584)            // +896*4           = 151040
#define OFF_KNNL   (151040+4096)            // +1024*4          = 155136
#define OFF_NX     (155136+512)             // +128*4           = 155648
#define SMEM_BYTES 155712                   // x2 = 311KB <= 320KB => 2 blk/WGP

// Workspace layout (bytes)
#define WS_KNN     0                        // B*S*K int  = 2,097,152
#define WS_NEWXYZ  2097152                  // B*S*3 f32  =   196,608
#define WS_W1T     2293760                  // 96*72  f16 =    13,824
#define WS_W2T     2307584                  // 64*136 f16 =    17,408
#define WS_W3T     2324992                  // 128*264 f16=    67,584
#define WS_SB      2392576                  // 896 f32    =     3,584

typedef __attribute__((ext_vector_type(16))) _Float16 v16h;
typedef __attribute__((ext_vector_type(8)))  _Float16 v8h;
typedef __attribute__((ext_vector_type(8)))  float    v8f;

union V16 { v16h v; v8h h[2]; };

// ---------------------------------------------------------------------------
// Kernel 1: fold BN into scale/bias, transpose+pad weights to f16 (B-operand
// layout: in-channel-major rows, out-channel cols, padded strides).
// ---------------------------------------------------------------------------
__global__ void prep_kernel(const float* __restrict__ w1, const float* __restrict__ w2,
                            const float* __restrict__ w3,
                            const float* __restrict__ g1, const float* __restrict__ g2,
                            const float* __restrict__ g3,
                            const float* __restrict__ be1, const float* __restrict__ be2,
                            const float* __restrict__ be3,
                            const float* __restrict__ me1, const float* __restrict__ me2,
                            const float* __restrict__ me3,
                            const float* __restrict__ va1, const float* __restrict__ va2,
                            const float* __restrict__ va3,
                            _Float16* __restrict__ w1t, _Float16* __restrict__ w2t,
                            _Float16* __restrict__ w3t, float* __restrict__ sb)
{
    int i = blockIdx.x * 256 + threadIdx.x;
    if (i < 96 * ST_W1) {                       // W1^T: rows = in(96, 67 real), cols = out(64)
        int r = i / ST_W1, c = i - r * ST_W1;
        float v = (r < 67 && c < 64) ? w1[c * 67 + r] : 0.f;
        w1t[i] = (_Float16)v;
    }
    int j = i - 96 * ST_W1;
    if (j >= 0 && j < 64 * ST_W2) {             // W2^T: rows = in(64), cols = out(128)
        int r = j / ST_W2, c = j - r * ST_W2;
        float v = (c < 128) ? w2[c * 64 + r] : 0.f;
        w2t[j] = (_Float16)v;
    }
    int k = j - 64 * ST_W2;
    if (k >= 0 && k < 128 * ST_W3) {            // W3^T: rows = in(128), cols = out(256)
        int r = k / ST_W3, c = k - r * ST_W3;
        float v = (c < 256) ? w3[c * 128 + r] : 0.f;
        w3t[k] = (_Float16)v;
    }
    int m = k - 128 * ST_W3;
    if (m >= 0 && m < 448) {                    // s1[64] b1[64] s2[128] b2[128] s3[256] b3[256]
        const float *g, *bb, *mm, *vv; int ch, so, bo;
        if (m < 64)       { g = g1; bb = be1; mm = me1; vv = va1; ch = m;       so = 0;   bo = 64;  }
        else if (m < 192) { g = g2; bb = be2; mm = me2; vv = va2; ch = m - 64;  so = 128; bo = 256; }
        else              { g = g3; bb = be3; mm = me3; vv = va3; ch = m - 192; so = 384; bo = 640; }
        float sc = g[ch] * rsqrtf(vv[ch] + 1e-5f);
        sb[so + ch] = sc;
        sb[bo + ch] = bb[ch] - mm[ch] * sc;
    }
}

// ---------------------------------------------------------------------------
// Kernel 2: farthest point sampling. One block per batch, 1024 threads,
// 8 points/thread resident in registers. Packed (distBits<<32)|~idx keys,
// per-wave shuffle reduce -> warr[32] -> wave-0 shuffle reduce. 2 barriers
// per serial step (this is the latency-critical serial chain of the op).
// ---------------------------------------------------------------------------
__device__ __forceinline__ unsigned long long shflx64(unsigned long long v, int mask)
{
    unsigned lo = (unsigned)v, hi = (unsigned)(v >> 32);
    lo = (unsigned)__shfl_xor((int)lo, mask, 32);
    hi = (unsigned)__shfl_xor((int)hi, mask, 32);
    return ((unsigned long long)hi << 32) | lo;
}

__global__ void fps_kernel(const float* __restrict__ xyz,
                           float* __restrict__ newxyz, float* __restrict__ out1)
{
    __shared__ float cur[3];
    __shared__ unsigned long long warr[32];
    const int b = blockIdx.x;
    const int tid = threadIdx.x;
    const int wave = tid >> 5, lane = tid & 31;
    const float* xb = xyz + (size_t)b * 3 * NPTS;

    float px[8], py[8], pz[8], dd[8];
#pragma unroll
    for (int t = 0; t < 8; t++) {
        int jj = tid + t * 1024;
        px[t] = xb[jj]; py[t] = xb[NPTS + jj]; pz[t] = xb[2 * NPTS + jj];
        dd[t] = 1e10f;
    }
    if (tid == 0) {
        float x = xb[0], y = xb[NPTS], z = xb[2 * NPTS];
        cur[0] = x; cur[1] = y; cur[2] = z;
        size_t r = (size_t)b * SCENT;
        newxyz[r * 3 + 0] = x; newxyz[r * 3 + 1] = y; newxyz[r * 3 + 2] = z;
        size_t o = (size_t)b * 3 * SCENT;
        out1[o] = x; out1[o + SCENT] = y; out1[o + 2 * SCENT] = z;
    }
    __syncthreads();

    for (int s = 1; s < SCENT; s++) {
        float cx = cur[0], cy = cur[1], cz = cur[2];
        unsigned long long key = 0ull;
#pragma unroll
        for (int t = 0; t < 8; t++) {
            float dx = px[t] - cx, dy = py[t] - cy, dz = pz[t] - cz;
            float d = dx * dx + dy * dy + dz * dz;
            dd[t] = fminf(dd[t], d);
            unsigned long long kk = ((unsigned long long)__float_as_uint(dd[t]) << 32)
                                  | (unsigned)(0xFFFFFFFFu - (unsigned)(tid + t * 1024));
            key = kk > key ? kk : key;
        }
#pragma unroll
        for (int off = 16; off > 0; off >>= 1) {
            unsigned long long o = shflx64(key, off);
            key = o > key ? o : key;
        }
        if (lane == 0) warr[wave] = key;
        __syncthreads();
        if (wave == 0) {
            unsigned long long k2 = warr[lane];
#pragma unroll
            for (int off = 16; off > 0; off >>= 1) {
                unsigned long long o = shflx64(k2, off);
                k2 = o > k2 ? o : k2;
            }
            if (lane == 0) {
                unsigned idx = 0xFFFFFFFFu - (unsigned)(k2 & 0xFFFFFFFFull);
                float x = xb[idx], y = xb[NPTS + idx], z = xb[2 * NPTS + idx];
                cur[0] = x; cur[1] = y; cur[2] = z;
                size_t r = (size_t)b * SCENT + s;
                newxyz[r * 3 + 0] = x; newxyz[r * 3 + 1] = y; newxyz[r * 3 + 2] = z;
                size_t o = (size_t)b * 3 * SCENT + s;
                out1[o] = x; out1[o + SCENT] = y; out1[o + 2 * SCENT] = z;
            }
        }
        __syncthreads();
    }
}

// ---------------------------------------------------------------------------
// Kernel 3: KNN. 1 thread per centroid, LDS-tiled points, register top-32
// maintained with fully-unrolled replace-max (no dynamic indexing => no
// scratch). Prefetch next tile while scanning current one.
// ---------------------------------------------------------------------------
__global__ void knn_kernel(const float* __restrict__ xyz,
                           const float* __restrict__ newxyz, int* __restrict__ knn)
{
    __shared__ float ptx[256], pty[256], ptz[256];
    int tid = threadIdx.x;
    int gid = blockIdx.x * 256 + tid;          // b*S + s
    int b = gid >> 11;
    const float* xb = xyz + (size_t)b * 3 * NPTS;
    float cx = newxyz[(size_t)gid * 3 + 0];
    float cy = newxyz[(size_t)gid * 3 + 1];
    float cz = newxyz[(size_t)gid * 3 + 2];

    float dA[KNB]; int iA[KNB];
#pragma unroll
    for (int k = 0; k < KNB; k++) { dA[k] = 3e38f; iA[k] = 0; }
    float worst = 3e38f; int wslot = 0;

    for (int jt = 0; jt < NPTS; jt += 256) {
        __syncthreads();
        ptx[tid] = xb[jt + tid];
        pty[tid] = xb[NPTS + jt + tid];
        ptz[tid] = xb[2 * NPTS + jt + tid];
        __syncthreads();
        if (jt + 256 < NPTS) {                 // prefetch next tile into caches
            __builtin_prefetch(xb + jt + 256 + tid, 0, 1);
            __builtin_prefetch(xb + NPTS + jt + 256 + tid, 0, 1);
            __builtin_prefetch(xb + 2 * NPTS + jt + 256 + tid, 0, 1);
        }
        for (int jj = 0; jj < 256; jj++) {
            float dx = ptx[jj] - cx, dy = pty[jj] - cy, dz = ptz[jj] - cz;
            float d = dx * dx + dy * dy + dz * dz;
            if (d < worst) {
#pragma unroll
                for (int k = 0; k < KNB; k++)
                    if (k == wslot) { dA[k] = d; iA[k] = jt + jj; }
                worst = -1e30f;
#pragma unroll
                for (int k = 0; k < KNB; k++)
                    if (dA[k] > worst) { worst = dA[k]; wslot = k; }
            }
        }
    }
    int* out = knn + (size_t)gid * KNB;
#pragma unroll
    for (int k = 0; k < KNB; k++) out[k] = iA[k];
}

// ---------------------------------------------------------------------------
// Async memory->LDS copy (gfx1250 GLOBAL_LOAD_ASYNC_TO_LDS_B128, ASYNCcnt).
// Per-lane: 16B from global addr -> LDS byte offset. No VGPR round-trip.
// ---------------------------------------------------------------------------
__device__ __forceinline__ void async_copy_b128(unsigned lds_off, const void* gaddr)
{
    asm volatile("global_load_async_to_lds_b128 %0, %1, off"
                 :: "v"(lds_off), "v"(gaddr) : "memory");
}
__device__ __forceinline__ void wait_asynccnt0()
{
    asm volatile("s_wait_asynccnt 0x0" ::: "memory");
}
__device__ __forceinline__ void lds_async_fill(unsigned lds_base, const void* src,
                                               int n16, int tid)
{
    const char* s = (const char*)src;
    for (int i = tid; i < n16; i += 256)
        async_copy_b128(lds_base + (unsigned)i * 16u, s + (size_t)i * 16u);
}

// ---------------------------------------------------------------------------
// WMMA GEMM helpers. With 8 waves and MT=8 row-tiles, wave w owns row-tile w:
// A fragments are loaded ONCE per layer and held in registers; the nt loop
// sweeps output-column tiles loading only B fragments (load-all-then-MMA so
// one dscnt wait covers the whole k-chain).
// A (16-bit 16x32): lane = M (mod 16); halves 0..7 = K {kb..kb+7},
//                   halves 8..15 = K {kb+16..kb+23}; kb = (lane>=16)*8.
// B (16-bit 32x16): lane = N (mod 16); halves = K in the same order as A.
// D (f32 16x16):    lane = N (mod 16); VGPR r = M r (+8 for lanes 16..31).
// ---------------------------------------------------------------------------
template<int KD, int STIN, int STW>
__device__ __forceinline__ void load_afrags(const _Float16* aIn, V16* af,
                                            int mt, int l15, int akb)
{
#pragma unroll
    for (int k = 0; k < KD / 32; k++) {
        const _Float16* ap = aIn + (mt * 16 + l15) * STIN + k * 32 + akb;
        af[k].h[0] = *(const v8h*)ap;
        af[k].h[1] = *(const v8h*)(ap + 16);
    }
}

template<int KD, int STIN, int OUTCH, int STW, int STOUT>
__device__ __forceinline__ void gemm_relu(const _Float16* aIn, const _Float16* wt,
                                          const float* scale, const float* bias,
                                          _Float16* out, int wave, int lane)
{
    constexpr int NK = KD / 32;
    const int l15 = lane & 15, hi = lane >> 4;
    const int akb  = hi * 8;
    const int krow = l15 + hi * 16;
    const int mt = wave;                      // fixed row-tile per wave (MT==8)
    V16 af[NK];
    load_afrags<KD, STIN, STW>(aIn, af, mt, l15, akb);
    for (int nt = 0; nt < OUTCH / 16; nt++) {
        V16 bf[NK];
#pragma unroll
        for (int k = 0; k < NK; k++) {
            const _Float16* bp = wt + (k * 32 + akb) * STW + nt * 16 + l15;
#pragma unroll
            for (int s = 0; s < 8; s++) {
                bf[k].h[0][s] = bp[s * STW];
                bf[k].h[1][s] = bp[(16 + s) * STW];
            }
        }
        v8f acc = {};
#pragma unroll
        for (int k = 0; k < NK; k++)
            acc = __builtin_amdgcn_wmma_f32_16x16x32_f16(false, af[k].v, false, bf[k].v,
                                                         (short)0, acc, false, false);
        int n = nt * 16 + l15;
        float sc = scale[n], bi = bias[n];
#pragma unroll
        for (int r = 0; r < 8; r++) {
            float v = acc[r] * sc + bi;
            v = v > 0.f ? v : 0.f;
            out[(mt * 16 + r + hi * 8) * STOUT + n] = (_Float16)v;
        }
    }
}

template<int KD, int STIN, int OUTCH, int STW>
__device__ __forceinline__ void gemm_pool(const _Float16* aIn, const _Float16* wt,
                                          const float* scale, const float* bias,
                                          int* pooled, int wave, int lane)
{
    constexpr int NK = KD / 32;
    const int l15 = lane & 15, hi = lane >> 4;
    const int akb  = hi * 8;
    const int krow = l15 + hi * 16;
    const int mt = wave;
    V16 af[NK];
    load_afrags<KD, STIN, STW>(aIn, af, mt, l15, akb);
    for (int nt = 0; nt < OUTCH / 16; nt++) {
        V16 bf[NK];
#pragma unroll
        for (int k = 0; k < NK; k++) {
            const _Float16* bp = wt + (k * 32 + akb) * STW + nt * 16 + l15;
#pragma unroll
            for (int s = 0; s < 8; s++) {
                bf[k].h[0][s] = bp[s * STW];
                bf[k].h[1][s] = bp[(16 + s) * STW];
            }
        }
        v8f acc = {};
#pragma unroll
        for (int k = 0; k < NK; k++)
            acc = __builtin_amdgcn_wmma_f32_16x16x32_f16(false, af[k].v, false, bf[k].v,
                                                         (short)0, acc, false, false);
        int n = nt * 16 + l15;
        float sc = scale[n], bi = bias[n];
        float mx = 0.f;                       // ReLU output >= 0
#pragma unroll
        for (int r = 0; r < 8; r++) {
            float v = acc[r] * sc + bi;
            v = v > 0.f ? v : 0.f;
            mx = fmaxf(mx, v);
        }
        // rows of tile mt belong to centroid mt/2 (K=32 rows per centroid);
        // nonneg float bits are order-preserving => int atomicMax == float max.
        atomicMax(&pooled[(mt >> 1) * 256 + n], __float_as_int(mx));
    }
}

// ---------------------------------------------------------------------------
// Kernel 4: fused gather + 3-layer MLP (f16 WMMA, f32 accum) + max-pool.
// Block = 4 centroids = 128 rows, 256 threads = 8 waves, 2 blocks/WGP.
// ---------------------------------------------------------------------------
__global__ void fused_kernel(const float* __restrict__ xyz, const float* __restrict__ pts,
                             const int* __restrict__ knn, const float* __restrict__ newxyz,
                             const _Float16* __restrict__ w1t, const _Float16* __restrict__ w2t,
                             const _Float16* __restrict__ w3t, const float* __restrict__ sbg,
                             float* __restrict__ out2)
{
    extern __shared__ char smem[];
    _Float16* bufX  = (_Float16*)(smem + OFF_BUFX);
    _Float16* bufY  = (_Float16*)(smem + OFF_BUFY);
    _Float16* bufZ  = (_Float16*)(smem + OFF_BUFZ);
    _Float16* wbuf  = (_Float16*)(smem + OFF_WBUF);
    float*    sb    = (float*)   (smem + OFF_SB);
    int*      pooled= (int*)     (smem + OFF_POOL);
    int*      knn_l = (int*)     (smem + OFF_KNNL);
    float*    nx    = (float*)   (smem + OFF_NX);

    const int tid = threadIdx.x, wave = tid >> 5, lane = tid & 31;
    const int g = blockIdx.x;
    const int b  = g >> 9;                    // 512 blocks per batch (2048/4)
    const int s0 = (g & 511) << 2;
    const size_t base = (size_t)b * SCENT + s0;

    // stage 0: W1^T async into wbuf; indices, centroids, scale/bias, pool init
    lds_async_fill(OFF_WBUF, w1t, (96 * ST_W1 * 2) / 16, tid);
    if (tid < 128) knn_l[tid] = knn[base * KNB + tid];
    if (tid < 12)  nx[tid] = newxyz[base * 3 + tid];
    for (int i = tid; i < 896;  i += 256) sb[i] = sbg[i];
    for (int i = tid; i < 1024; i += 256) pooled[i] = 0;
    wait_asynccnt0();
    __syncthreads();

    // stage 1: gather [xyz - centroid | point feats | zero pad] as f16
    const float* xb = xyz + (size_t)b * 3 * NPTS;
    const float* pb = pts + (size_t)b * CIN * NPTS;
    for (int e = tid; e < 128 * 96; e += 256) {
        int row = e / 96, col = e - row * 96;
        int idx = knn_l[row];
        float v;
        if (col < 3)       v = xb[col * NPTS + idx] - nx[(row >> 5) * 3 + col];
        else if (col < 67) v = pb[(col - 3) * NPTS + idx];
        else               v = 0.f;
        bufX[row * ST_X + col] = (_Float16)v;
    }
    __syncthreads();

    // layer 1: (128 x 96) x (96 x 64)
    gemm_relu<96, ST_X, C1, ST_W1, ST_Y>(bufX, wbuf, sb + 0, sb + 64, bufY, wave, lane);
    __syncthreads();
    lds_async_fill(OFF_WBUF, w2t, (64 * ST_W2 * 2) / 16, tid);
    wait_asynccnt0();
    __syncthreads();

    // layer 2: (128 x 64) x (64 x 128)
    gemm_relu<64, ST_Y, C2, ST_W2, ST_Z>(bufY, wbuf, sb + 128, sb + 256, bufZ, wave, lane);
    __syncthreads();
    lds_async_fill(OFF_WBUF, w3t, (128 * ST_W3 * 2) / 16, tid);
    wait_asynccnt0();
    __syncthreads();

    // layer 3: (128 x 128) x (128 x 256) + max-pool over K=32 rows
    gemm_pool<128, ST_Z, C3, ST_W3>(bufZ, wbuf, sb + 384, sb + 640, pooled, wave, lane);
    __syncthreads();

    // write pooled (B,256,S) slice
    for (int i = tid; i < 1024; i += 256) {
        int cl = i >> 8, ch = i & 255;
        out2[((size_t)b * C3 + ch) * SCENT + (s0 + cl)] = __int_as_float(pooled[i]);
    }
}

// ---------------------------------------------------------------------------
// Host launcher
// ---------------------------------------------------------------------------
extern "C" void kernel_launch(void* const* d_in, const int* in_sizes, int n_in,
                              void* d_out, int out_size, void* d_ws, size_t ws_size,
                              hipStream_t stream)
{
    const float* xyz = (const float*)d_in[0];
    const float* pts = (const float*)d_in[1];
    const float* w1  = (const float*)d_in[2];
    const float* w2  = (const float*)d_in[3];
    const float* w3  = (const float*)d_in[4];
    const float* g1  = (const float*)d_in[5];
    const float* g2  = (const float*)d_in[6];
    const float* g3  = (const float*)d_in[7];
    const float* be1 = (const float*)d_in[8];
    const float* be2 = (const float*)d_in[9];
    const float* be3 = (const float*)d_in[10];
    const float* me1 = (const float*)d_in[11];
    const float* me2 = (const float*)d_in[12];
    const float* me3 = (const float*)d_in[13];
    const float* va1 = (const float*)d_in[14];
    const float* va2 = (const float*)d_in[15];
    const float* va3 = (const float*)d_in[16];

    char* ws = (char*)d_ws;
    int*      knn    = (int*)     (ws + WS_KNN);
    float*    newxyz = (float*)   (ws + WS_NEWXYZ);
    _Float16* w1t    = (_Float16*)(ws + WS_W1T);
    _Float16* w2t    = (_Float16*)(ws + WS_W2T);
    _Float16* w3t    = (_Float16*)(ws + WS_W3T);
    float*    sb     = (float*)   (ws + WS_SB);

    float* out1 = (float*)d_out;                      // (B,3,S)
    float* out2 = out1 + (size_t)BATCH * 3 * SCENT;   // (B,256,S)

    prep_kernel<<<195, 256, 0, stream>>>(w1, w2, w3, g1, g2, g3, be1, be2, be3,
                                         me1, me2, me3, va1, va2, va3,
                                         w1t, w2t, w3t, sb);
    fps_kernel<<<BATCH, 1024, 0, stream>>>(xyz, newxyz, out1);
    knn_kernel<<<(BATCH * SCENT) / 256, 256, 0, stream>>>(xyz, newxyz, knn);
    fused_kernel<<<(BATCH * SCENT) / 4, 256, SMEM_BYTES, stream>>>(
        xyz, pts, knn, newxyz, w1t, w2t, w3t, sb, out2);
}
